// MaskAttention_27015344292152
// MI455X (gfx1250) — matrix operand order repaired and checked by reading, not compiled, not run on hardware
//
#include <hip/hip_runtime.h>
#include <hip/hip_bf16.h>
#include <stdint.h>

typedef _Float16 v16h __attribute__((ext_vector_type(16)));
typedef _Float16 v8h  __attribute__((ext_vector_type(8)));
typedef float    v8f  __attribute__((ext_vector_type(8)));
typedef unsigned int u32x4 __attribute__((ext_vector_type(4)));
typedef int          i32x4 __attribute__((ext_vector_type(4)));
typedef int          i32x8 __attribute__((ext_vector_type(8)));

// ---- problem constants ----
constexpr int BSZ   = 8;
constexpr int NTOK  = 1024;
constexpr int DIMC  = 512;
constexpr int HEADS = 8;
constexpr int DH    = 64;
constexpr int INNER = 512;   // HEADS*DH
constexpr int NSEG  = 4;
constexpr int SEGL  = 256;
constexpr float QSCALE = 0.125f;   // 64^-0.5

#if defined(__has_builtin)
#if __has_builtin(__builtin_amdgcn_tensor_load_to_lds)
#define HAVE_TDM 1
#endif
#endif

// ---------------- WMMA helpers (gfx1250 wave32) ----------------
__device__ inline v8f wmma_f16(v16h a, v16h b, v8f c) {
    return __builtin_amdgcn_wmma_f32_16x16x32_f16(
        false, a, false, b, (short)0, c, false, false);
}

__device__ inline v16h cat8(v8h lo, v8h hi) {
    v16h r;
#pragma unroll
    for (int i = 0; i < 8; ++i) { r[i] = lo[i]; r[i + 8] = hi[i]; }
    return r;
}

// A-matrix 16x32 fragment: lane m(0..15)=row m; lanes 0-15 K=[0..7,16..23],
// lanes 16-31 K=[8..15,24..31]; base at (row0,k0), ld in elements.
template <typename P>
__device__ inline v16h load_a16x32(const P* base, int ld, int lane) {
    int m = lane & 15, hh = lane >> 4;
    const v8h* p0 = (const v8h*)((const _Float16*)base + (size_t)m * ld + 8 * hh);
    const v8h* p1 = (const v8h*)((const _Float16*)base + (size_t)m * ld + 8 * hh + 16);
    return cat8(*p0, *p1);
}

// B-matrix 32x16 fragment: lane n(0..15)=col n; lanes 0-15 K=[0..7,16..23],
// lanes 16-31 K=[8..15,24..31]; base at (col0,k0) of K-major [col][k] array.
template <typename P>
__device__ inline v16h load_b32x16(const P* base, int ld, int lane) {
    int n = lane & 15, hh = lane >> 4;
    const v8h* p0 = (const v8h*)((const _Float16*)base + (size_t)n * ld + 8 * hh);
    const v8h* p1 = (const v8h*)((const _Float16*)base + (size_t)n * ld + 8 * hh + 16);
    return cat8(*p0, *p1);
}

// ---------------- gfx1250 data movers ----------------
// TDM: load a rows x tile_k (f16) tile from a row_stride-pitched tensor into LDS.
__device__ inline void tdm_load_tile_f16(void* lds, const void* gptr,
                                         unsigned rows, unsigned tile_k,
                                         unsigned tensor_k, unsigned row_stride) {
#ifdef HAVE_TDM
    unsigned lo = (unsigned)(uintptr_t)lds;                 // LDS byte offset
    unsigned long long ga = (unsigned long long)(uintptr_t)gptr;
    u32x4 g0;
    g0[0] = 1u;                                             // count=1, user mode
    g0[1] = lo;                                             // lds_addr
    g0[2] = (unsigned)(ga & 0xffffffffu);                   // global_addr[31:0]
    g0[3] = (unsigned)((ga >> 32) & 0x1ffffffu) | (2u << 30); // addr[56:32], type=2
    i32x8 g1;
    g1[0] = 0x00010000;                                     // data_size=1 (2 bytes)
    g1[1] = (int)((tensor_k & 0xffffu) << 16);              // tensor_dim0[15:0]
    g1[2] = (int)(((tensor_k >> 16) & 0xffffu) | ((rows & 0xffffu) << 16)); // td0 hi | td1 lo
    g1[3] = (int)(((rows >> 16) & 0xffffu) | ((tile_k & 0xffffu) << 16));   // td1 hi | tile_dim0
    g1[4] = (int)(rows & 0xffffu);                          // tile_dim1 | tile_dim2=0
    g1[5] = (int)row_stride;                                // tensor_dim0_stride[31:0]
    g1[6] = 0;
    g1[7] = 0;
    i32x4 z4 = (i32x4)0;
#if __clang_major__ >= 23
    i32x8 z8 = (i32x8)0;
    __builtin_amdgcn_tensor_load_to_lds(g0, g1, z4, z4, z8, 0);
#else
    __builtin_amdgcn_tensor_load_to_lds(g0, g1, z4, z4, 0);
#endif
#endif
}

__device__ inline void tdm_wait0() {
#if defined(__has_builtin) && __has_builtin(__builtin_amdgcn_s_wait_tensorcnt)
    __builtin_amdgcn_s_wait_tensorcnt((unsigned short)0);
#else
    asm volatile("s_wait_tensorcnt 0x0" ::: "memory");
#endif
}

// Async 16B global->LDS copy (ASYNCcnt path).
__device__ inline void async_copy_b128(void* lds_dst, const void* g) {
    unsigned ldsoff = (unsigned)(uintptr_t)lds_dst;
    unsigned long long ga = (unsigned long long)(uintptr_t)g;
    asm volatile("global_load_async_to_lds_b128 %0, %1, off"
                 :: "v"(ldsoff), "v"(ga) : "memory");
}
__device__ inline void async_wait0() {
#if defined(__has_builtin) && __has_builtin(__builtin_amdgcn_s_wait_asynccnt)
    __builtin_amdgcn_s_wait_asynccnt(0);
#else
    asm volatile("s_wait_asynccnt 0x0" ::: "memory");
#endif
}

// Shared WMMA micro-kernel over one staged 64x256 B panel (grouped loads).
__device__ inline void panel_mma(const _Float16* __restrict__ arow,   // (row0, kbase)
                                 int lda,
                                 const _Float16* __restrict__ bp,     // LDS panel
                                 int lane, v8f acc[4]) {
#pragma unroll 4
    for (int k0 = 0; k0 < 256; k0 += 32) {
        // prefetch next A chunk into L0/L2 while this step computes
        if (k0 + 32 < 256)
            __builtin_prefetch((const _Float16*)arow + (lane & 15) * lda + k0 + 32, 0, 1);
        v16h a  = load_a16x32(arow + k0, lda, lane);
        v16h b0 = load_b32x16(bp + (0 * 16) * 256 + k0, 256, lane);
        v16h b1 = load_b32x16(bp + (1 * 16) * 256 + k0, 256, lane);
        v16h b2 = load_b32x16(bp + (2 * 16) * 256 + k0, 256, lane);
        v16h b3 = load_b32x16(bp + (3 * 16) * 256 + k0, 256, lane);
        acc[0] = wmma_f16(a, b0, acc[0]);
        acc[1] = wmma_f16(a, b1, acc[1]);
        acc[2] = wmma_f16(a, b2, acc[2]);
        acc[3] = wmma_f16(a, b3, acc[3]);
    }
}

// ---------------- Kernel 1: LayerNorm -> f16 ----------------
__global__ __launch_bounds__(128) void ln_kernel(
    const float* __restrict__ x, const float* __restrict__ gamma,
    const float* __restrict__ beta, _Float16* __restrict__ h16) {
    int row = blockIdx.x;
    int tid = threadIdx.x;
    const float* xr = x + (size_t)row * DIMC;
    float4 v = ((const float4*)xr)[tid];
    float s  = v.x + v.y + v.z + v.w;
    float s2 = v.x * v.x + v.y * v.y + v.z * v.z + v.w * v.w;
#pragma unroll
    for (int off = 1; off < 32; off <<= 1) {
        s  += __shfl_xor(s,  off, 32);
        s2 += __shfl_xor(s2, off, 32);
    }
    __shared__ float red[2][4];
    int w = tid >> 5;
    if ((tid & 31) == 0) { red[0][w] = s; red[1][w] = s2; }
    __syncthreads();
    s  = red[0][0] + red[0][1] + red[0][2] + red[0][3];
    s2 = red[1][0] + red[1][1] + red[1][2] + red[1][3];
    float mu   = s * (1.f / 512.f);
    float var  = s2 * (1.f / 512.f) - mu * mu;
    float rstd = rsqrtf(var + 1e-5f);
    int i = tid * 4;
    _Float16* hr = h16 + (size_t)row * DIMC + i;
    hr[0] = (_Float16)((v.x - mu) * rstd * gamma[i + 0] + beta[i + 0]);
    hr[1] = (_Float16)((v.y - mu) * rstd * gamma[i + 1] + beta[i + 1]);
    hr[2] = (_Float16)((v.z - mu) * rstd * gamma[i + 2] + beta[i + 2]);
    hr[3] = (_Float16)((v.w - mu) * rstd * gamma[i + 3] + beta[i + 3]);
}

// ---------------- Kernel 2: weight transpose + f16 convert ----------------
__global__ __launch_bounds__(256) void cvt_wqkv(const float* __restrict__ w,
                                                _Float16* __restrict__ wT) {
    int idx = blockIdx.x * 256 + threadIdx.x;     // wT[c][k], 1536*512
    int c = idx / DIMC, k = idx % DIMC;
    wT[idx] = (_Float16)w[(size_t)k * (3 * INNER) + c];
}
__global__ __launch_bounds__(256) void cvt_wout(const float* __restrict__ w,
                                                _Float16* __restrict__ wT) {
    int idx = blockIdx.x * 256 + threadIdx.x;     // wT[c][k], 512*512
    int c = idx / INNER, k = idx % INNER;
    wT[idx] = (_Float16)w[(size_t)k * DIMC + c];
}

// ------ Kernel 3: QKV GEMM, B panel double-buffered via TDM (TENSORcnt) -----
// Block: 256 thr = 8 waves -> 128x64 tile; LDS: 2 x (64 cols x 256 K) = 64 KB.
// Half k+1's DMA overlaps half k's WMMA loop.
__global__ __launch_bounds__(256) void qkv_gemm(
    const _Float16* __restrict__ h16, const _Float16* __restrict__ wqkvT,
    _Float16* __restrict__ q16, _Float16* __restrict__ k16,
    _Float16* __restrict__ vT) {
    __shared__ _Float16 bpanel[2][64 * 256];      // 2 x 32 KB
    int w = threadIdx.x >> 5, lane = threadIdx.x & 31;
    const int colTiles = (3 * INNER) / 64;        // 24
    int ct = blockIdx.x % colTiles, rb = blockIdx.x / colTiles;
    int row0 = rb * 128 + w * 16;                 // row over B*N
    int n0   = ct * 64;
    const _Float16* gpanel = wqkvT + (size_t)n0 * DIMC;
    v8f acc[4] = {};

#ifdef HAVE_TDM
    if (threadIdx.x < 32) {
        tdm_load_tile_f16(bpanel[0], gpanel + 0, 64, 256, DIMC, DIMC);
        tdm_wait0();
    }
    __syncthreads();
    // overlap: DMA half 1 while computing half 0
    if (threadIdx.x < 32)
        tdm_load_tile_f16(bpanel[1], gpanel + 256, 64, 256, DIMC, DIMC);
    panel_mma(h16 + (size_t)row0 * DIMC + 0, DIMC, bpanel[0], lane, acc);
    if (threadIdx.x < 32) tdm_wait0();
    __syncthreads();
    panel_mma(h16 + (size_t)row0 * DIMC + 256, DIMC, bpanel[1], lane, acc);
#else
    for (int kh = 0; kh < 2; ++kh) {
        for (int i = threadIdx.x; i < 64 * 256 / 8; i += 256) {
            int r = i >> 5, kc = i & 31;
            *(v8h*)&bpanel[0][r * 256 + kc * 8] =
                *(const v8h*)(gpanel + (size_t)r * DIMC + kh * 256 + kc * 8);
        }
        __syncthreads();
        panel_mma(h16 + (size_t)row0 * DIMC + kh * 256, DIMC, bpanel[0], lane, acc);
        __syncthreads();
    }
#endif

    int b = row0 / NTOK, nbase = row0 % NTOK;
    int rowoff = (lane >> 4) * 8, cl = lane & 15;
#pragma unroll
    for (int t = 0; t < 4; ++t) {
        int col = n0 + t * 16 + cl;
#pragma unroll
        for (int r = 0; r < 8; ++r) {
            int n = nbase + rowoff + r;
            float v = acc[t][r];
            if (col < INNER) {                    // Q (pre-scaled)
                int hh = col >> 6, d = col & 63;
                q16[(((size_t)b * HEADS + hh) * NTOK + n) * DH + d] = (_Float16)(v * QSCALE);
            } else if (col < 2 * INNER) {         // K row-major [n][d]
                int c = col - INNER; int hh = c >> 6, d = c & 63;
                k16[(((size_t)b * HEADS + hh) * NTOK + n) * DH + d] = (_Float16)v;
            } else {                              // V transposed [d][n]
                int c = col - 2 * INNER; int hh = c >> 6, d = c & 63;
                vT[(((size_t)b * HEADS + hh) * DH + d) * NTOK + n] = (_Float16)v;
            }
        }
    }
}

// ---------------- Kernel 4: fused masked flash attention (per-wave) ----------
__global__ __launch_bounds__(256) void attn_kernel(
    const _Float16* __restrict__ q16, const _Float16* __restrict__ k16,
    const _Float16* __restrict__ vT, const int* __restrict__ m_labels,
    _Float16* __restrict__ o16) {
    __shared__ _Float16 pbuf[8][16 * 32];
    int w = threadIdx.x >> 5, lane = threadIdx.x & 31;
    int gw = blockIdx.x * 8 + w;
    const int mtiles = NTOK / 16;
    int mt = gw % mtiles;
    int bh = gw / mtiles;
    int h  = bh % HEADS, b = bh / HEADS;
    int cl = lane & 15, rowoff = (lane >> 4) * 8;
    size_t obase = (size_t)b * NTOK * INNER + (size_t)h * DH;

    int rowseg = (mt * 16) / SEGL;
    if (m_labels[b * NSEG + rowseg] == 0) {
#pragma unroll
        for (int t = 0; t < 4; ++t)
#pragma unroll
            for (int r = 0; r < 8; ++r) {
                int n = mt * 16 + rowoff + r;
                o16[obase + (size_t)n * INNER + t * 16 + cl] = (_Float16)0.f;
            }
        return;
    }

    const _Float16* qbase = q16 + (((size_t)b * HEADS + h) * NTOK + mt * 16) * DH;
    v16h aq0 = load_a16x32(qbase + 0,  DH, lane);
    v16h aq1 = load_a16x32(qbase + 32, DH, lane);
    v8f o0 = {}, o1 = {}, o2 = {}, o3 = {};
    float mrow[8], lrow[8];
#pragma unroll
    for (int r = 0; r < 8; ++r) { mrow[r] = -3.0e38f; lrow[r] = 0.f; }

    const _Float16* kbase = k16 + ((size_t)b * HEADS + h) * NTOK * DH;
    const _Float16* vbase = vT  + ((size_t)b * HEADS + h) * DH * NTOK;
    _Float16* pb = pbuf[w];

    for (int seg = 0; seg < NSEG; ++seg) {
        if (m_labels[b * NSEG + seg] == 0) continue;
        for (int kk = seg * SEGL; kk < (seg + 1) * SEGL; kk += 32) {
            const _Float16* kb0 = kbase + (size_t)kk * DH;
            const _Float16* kb1 = kbase + (size_t)(kk + 16) * DH;
            v16h bk00 = load_b32x16(kb0 + 0,  DH, lane);
            v16h bk01 = load_b32x16(kb0 + 32, DH, lane);
            v16h bk10 = load_b32x16(kb1 + 0,  DH, lane);
            v16h bk11 = load_b32x16(kb1 + 32, DH, lane);
            v8f s0 = {}, s1 = {};
            s0 = wmma_f16(aq0, bk00, s0);
            s0 = wmma_f16(aq1, bk01, s0);
            s1 = wmma_f16(aq0, bk10, s1);
            s1 = wmma_f16(aq1, bk11, s1);
#pragma unroll
            for (int r = 0; r < 8; ++r) {
                float x = fmaxf(s0[r], s1[r]);
#pragma unroll
                for (int off = 1; off < 16; off <<= 1) x = fmaxf(x, __shfl_xor(x, off, 32));
                float mnew = fmaxf(mrow[r], x);
                float corr = __expf(mrow[r] - mnew);
                mrow[r] = mnew;
                lrow[r] *= corr;
                o0[r] *= corr; o1[r] *= corr; o2[r] *= corr; o3[r] *= corr;
                float p0 = __expf(s0[r] - mnew);
                float p1 = __expf(s1[r] - mnew);
                float ps = p0 + p1;
#pragma unroll
                for (int off = 1; off < 16; off <<= 1) ps += __shfl_xor(ps, off, 32);
                lrow[r] += ps;
                int rr = rowoff + r;
                pb[rr * 32 + cl]      = (_Float16)p0;
                pb[rr * 32 + 16 + cl] = (_Float16)p1;
            }
            v16h pa  = load_a16x32(pb, 32, lane);
            v16h bv0 = load_b32x16(vbase + (size_t)(0 * 16) * NTOK + kk, NTOK, lane);
            v16h bv1 = load_b32x16(vbase + (size_t)(1 * 16) * NTOK + kk, NTOK, lane);
            v16h bv2 = load_b32x16(vbase + (size_t)(2 * 16) * NTOK + kk, NTOK, lane);
            v16h bv3 = load_b32x16(vbase + (size_t)(3 * 16) * NTOK + kk, NTOK, lane);
            o0 = wmma_f16(pa, bv0, o0);
            o1 = wmma_f16(pa, bv1, o1);
            o2 = wmma_f16(pa, bv2, o2);
            o3 = wmma_f16(pa, bv3, o3);
        }
    }
#pragma unroll
    for (int r = 0; r < 8; ++r) {
        float inv = 1.0f / lrow[r];
        int n = mt * 16 + rowoff + r;
        size_t ob = obase + (size_t)n * INNER;
        o16[ob + 0 * 16 + cl] = (_Float16)(o0[r] * inv);
        o16[ob + 1 * 16 + cl] = (_Float16)(o1[r] * inv);
        o16[ob + 2 * 16 + cl] = (_Float16)(o2[r] * inv);
        o16[ob + 3 * 16 + cl] = (_Float16)(o3[r] * inv);
    }
}

// -- Kernel 5: output GEMM + bias, B double-buffered via async LDS (ASYNCcnt) --
__global__ __launch_bounds__(256) void out_gemm(
    const _Float16* __restrict__ o16, const _Float16* __restrict__ woutT,
    const float* __restrict__ b_out, float* __restrict__ out) {
    __shared__ _Float16 bpanel[2][64 * 256];      // 2 x 32 KB
    int w = threadIdx.x >> 5, lane = threadIdx.x & 31;
    const int colTiles = DIMC / 64;               // 8
    int ct = blockIdx.x % colTiles, rb = blockIdx.x / colTiles;
    int row0 = rb * 128 + w * 16;
    int n0   = ct * 64;
    const _Float16* gpanel = woutT + (size_t)n0 * INNER;
    v8f acc[4] = {};

    // stage half 0: 64 rows x 256 halfs = 2048 x 16B chunks, 8 per thread
#pragma unroll
    for (int j = 0; j < 8; ++j) {
        int idx = threadIdx.x + j * 256;
        int r = idx >> 5, kc = idx & 31;
        async_copy_b128(&bpanel[0][r * 256 + kc * 8],
                        gpanel + (size_t)r * INNER + kc * 8);
    }
    async_wait0();
    __syncthreads();
    // overlap: async-DMA half 1 while computing half 0
#pragma unroll
    for (int j = 0; j < 8; ++j) {
        int idx = threadIdx.x + j * 256;
        int r = idx >> 5, kc = idx & 31;
        async_copy_b128(&bpanel[1][r * 256 + kc * 8],
                        gpanel + (size_t)r * INNER + 256 + kc * 8);
    }
    panel_mma(o16 + (size_t)row0 * INNER + 0, INNER, bpanel[0], lane, acc);
    async_wait0();
    __syncthreads();
    panel_mma(o16 + (size_t)row0 * INNER + 256, INNER, bpanel[1], lane, acc);

    int rowoff = (lane >> 4) * 8, cl = lane & 15;
#pragma unroll
    for (int t = 0; t < 4; ++t) {
        int col = n0 + t * 16 + cl;
        float bb = b_out[col];
#pragma unroll
        for (int r = 0; r < 8; ++r) {
            int row = row0 + rowoff + r;
            out[(size_t)row * DIMC + col] = acc[t][r] + bb;
        }
    }
}

// ---------------- launch ----------------
extern "C" void kernel_launch(void* const* d_in, const int* in_sizes, int n_in,
                              void* d_out, int out_size, void* d_ws, size_t ws_size,
                              hipStream_t stream) {
    const float* x        = (const float*)d_in[0];
    const int*   m_labels = (const int*)  d_in[1];
    const float* gamma    = (const float*)d_in[2];
    const float* beta     = (const float*)d_in[3];
    const float* w_qkv    = (const float*)d_in[4];
    const float* w_out    = (const float*)d_in[5];
    const float* b_out    = (const float*)d_in[6];
    float* out = (float*)d_out;

    _Float16* h16   = (_Float16*)d_ws;                        // [B*N, 512]
    _Float16* wqkvT = h16   + (size_t)BSZ * NTOK * DIMC;      // [1536, 512]
    _Float16* woutT = wqkvT + (size_t)3 * INNER * DIMC;       // [512, 512]
    _Float16* q16   = woutT + (size_t)DIMC * INNER;           // [B,H,N,Dh]
    _Float16* k16   = q16   + (size_t)BSZ * HEADS * NTOK * DH;
    _Float16* vT    = k16   + (size_t)BSZ * HEADS * NTOK * DH;  // [B,H,Dh,N]
    _Float16* o16   = vT    + (size_t)BSZ * HEADS * NTOK * DH;  // [B*N, 512]

    ln_kernel<<<BSZ * NTOK, 128, 0, stream>>>(x, gamma, beta, h16);
    cvt_wqkv<<<(3 * INNER * DIMC) / 256, 256, 0, stream>>>(w_qkv, wqkvT);
    cvt_wout<<<(DIMC * INNER) / 256, 256, 0, stream>>>(w_out, woutT);

    // 64 row-blocks (128 rows) x 24 col-tiles
    qkv_gemm<<<64 * 24, 256, 0, stream>>>(h16, wqkvT, q16, k16, vT);
    // B*H*(N/16) = 4096 waves, 8 per block
    attn_kernel<<<4096 / 8, 256, 0, stream>>>(q16, k16, vT, m_labels, o16);
    // 64 row-blocks x 8 col-tiles
    out_gemm<<<64 * 8, 256, 0, stream>>>(o16, woutT, b_out, out);
}
